// NGP_635655160022
// MI455X (gfx1250) — matrix operand order repaired and checked by reading, not compiled, not run on hardware
//
#include <hip/hip_runtime.h>
#include <hip/hip_bf16.h>

typedef __attribute__((ext_vector_type(16))) _Float16 v16h;
typedef __attribute__((ext_vector_type(8)))  float    v8f;

#define T_HASH   524288u     // hash table size (2^19)
#define NPB      256         // points per block (8 waves x 2 tiles x 16 pts)
#define LDA      68          // padded row stride (halfs) for activations: 136B -> bank-conflict-free
#define W1LD     36          // padded K-stride for dW1^T (72B)
#define WLD      68          // padded K-stride for 64-K weights (136B)

// dynamic LDS layout (bytes)
#define OFF_W1T   0
#define OFF_W2T   (OFF_W1T + 64 * W1LD * 2)        // 4608
#define OFF_C1T   (OFF_W2T + 16 * WLD * 2)         // +2176
#define OFF_C2T   (OFF_C1T + 64 * WLD * 2)         // +8704
#define OFF_C3T   (OFF_C2T + 64 * WLD * 2)         // +8704
#define OFF_B1    (OFF_C3T + 16 * WLD * 2)         // +2176
#define OFF_B2    (OFF_B1 + 64 * 4)
#define OFF_BC1   (OFF_B2 + 16 * 4)
#define OFF_BC2   (OFF_BC1 + 64 * 4)
#define OFF_BC3   (OFF_BC2 + 64 * 4)
#define OFF_ACT   (OFF_BC3 + 16 * 4)
#define OFF_CIN   (OFF_ACT + NPB * LDA * 2)
#define OFF_COL   (OFF_CIN + NPB * LDA * 2)
#define OFF_SIG   (OFF_COL + NPB * 3 * 4)
#define OFF_MASK  (OFF_SIG + NPB * 4)
#define SMEM_BYTES (OFF_MASK + NPB)                // ~100 KB (<< 320 KB/WG)

__device__ __constant__ float c_NL[16] = {
    16.f, 22.f, 31.f, 42.f, 58.f, 80.f, 111.f, 154.f,
    213.f, 294.f, 406.f, 562.f, 777.f, 1073.f, 1483.f, 2048.f};

// ---------- WMMA fragment helpers (CDNA5 ISA 7.12.2 layouts, wave32) ----------

// A-matrix 16x32 f16 from row-major LDS buf[16][ld] at K-offset k0.
// lanes 0-15: row M=lane, K 0..7 & 16..23 ; lanes 16-31: row M=lane-16, K 8..15 & 24..31
__device__ __forceinline__ v16h load_a_frag(const _Float16* buf, int ld, int k0) {
    const int lane = threadIdx.x & 31;
    const int m    = lane & 15;
    const int hi   = lane >> 4;
    const unsigned* r = (const unsigned*)(buf + m * ld + k0);  // 2 halfs per uint
    union { v16h h; unsigned u[8]; } A;
    const int b1 = hi * 4;       // uint idx of K pairs 0-3  -> K = hi*8 + 2p
    const int b2 = 8 + hi * 4;   // uint idx of K pairs 4-7  -> K = 16 + hi*8 + 2p
#pragma unroll
    for (int p = 0; p < 4; ++p) A.u[p]     = r[b1 + p];
#pragma unroll
    for (int p = 0; p < 4; ++p) A.u[4 + p] = r[b2 + p];
    return A.h;
}

// B-matrix 32x16 f16 from transposed weights wt[n][ldk] (wt = W^T), tile col n0, K-offset k0.
// lanes 0-15: col N=lane, K 0..7 & 16..23 ; lanes 16-31: col N=lane-16, K 8..15 & 24..31
__device__ __forceinline__ v16h load_b_frag(const _Float16* wt, int ldk, int n0, int k0) {
    const int lane = threadIdx.x & 31;
    const int n    = n0 + (lane & 15);
    const int kb   = (lane >> 4) * 8;
    const unsigned* c = (const unsigned*)(wt + n * ldk + k0 + kb);
    union { v16h h; unsigned u[8]; } B;
#pragma unroll
    for (int p = 0; p < 8; ++p) B.u[p] = c[p + (p >> 2) * 4];
    return B.h;
}

// C/D 16x16 f32: lane holds col N=lane&15; VGPR r holds row M = r + (lane>=16 ? 8 : 0)
__device__ __forceinline__ v8f bias_frag(const float* b, int n0) {
    const int lane = threadIdx.x & 31;
    const float v = b[n0 + (lane & 15)];
    v8f c;
#pragma unroll
    for (int r = 0; r < 8; ++r) c[r] = v;
    return c;
}

__device__ __forceinline__ void store_d_frag(_Float16* buf, int ld, int n0, v8f d, bool relu) {
    const int lane = threadIdx.x & 31;
    const int n  = n0 + (lane & 15);
    const int mb = (lane >> 4) * 8;
#pragma unroll
    for (int r = 0; r < 8; ++r) {
        float v = d[r];
        if (relu) v = fmaxf(v, 0.f);
        buf[(mb + r) * ld + n] = (_Float16)v;
    }
}

__device__ __forceinline__ v8f wmma32(v16h a, v16h b, v8f c) {
    return __builtin_amdgcn_wmma_f32_16x16x32_f16(false, a, false, b, (short)0, c, false, false);
}

// -----------------------------------------------------------------------------

__global__ __launch_bounds__(NPB) void ngp_fused_kernel(
    const float* __restrict__ x, const float* __restrict__ dir,
    const float* __restrict__ tables,
    const float* __restrict__ dW1, const float* __restrict__ db1,
    const float* __restrict__ dW2, const float* __restrict__ db2,
    const float* __restrict__ cW1, const float* __restrict__ cb1,
    const float* __restrict__ cW2, const float* __restrict__ cb2,
    const float* __restrict__ cW3, const float* __restrict__ cb3,
    float* __restrict__ out_color, float* __restrict__ out_sigma)
{
    extern __shared__ char smem[];
    _Float16* s_w1t = (_Float16*)(smem + OFF_W1T);   // dW1^T [64][W1LD]
    _Float16* s_w2t = (_Float16*)(smem + OFF_W2T);   // dW2^T [16][WLD]
    _Float16* s_c1t = (_Float16*)(smem + OFF_C1T);   // cW1^T [64][WLD], K 43..63 zero
    _Float16* s_c2t = (_Float16*)(smem + OFF_C2T);   // cW2^T [64][WLD]
    _Float16* s_c3t = (_Float16*)(smem + OFF_C3T);   // cW3^T [16][WLD], n 3..15 zero
    float* s_b1  = (float*)(smem + OFF_B1);
    float* s_b2  = (float*)(smem + OFF_B2);
    float* s_bc1 = (float*)(smem + OFF_BC1);
    float* s_bc2 = (float*)(smem + OFF_BC2);
    float* s_bc3 = (float*)(smem + OFF_BC3);
    _Float16* s_act = (_Float16*)(smem + OFF_ACT);   // feat / hidden activations
    _Float16* s_cin = (_Float16*)(smem + OFF_CIN);   // [hden16 | posenc27 | zeros]
    float* s_col = (float*)(smem + OFF_COL);
    float* s_sig = (float*)(smem + OFF_SIG);
    unsigned char* s_mask = (unsigned char*)(smem + OFF_MASK);

    const int tid = threadIdx.x;
    const int p0  = blockIdx.x * NPB;

    // ---------------- Phase 1: stage weights (f32 -> f16, transposed, padded) ----------------
    for (int i = tid; i < 64 * 32; i += NPB) { int n = i >> 5, k = i & 31; s_w1t[n * W1LD + k] = (_Float16)dW1[k * 64 + n]; }
    for (int i = tid; i < 16 * 64; i += NPB) { int n = i >> 6, k = i & 63; s_w2t[n * WLD + k] = (_Float16)dW2[k * 16 + n]; }
    for (int i = tid; i < 64 * 64; i += NPB) { int n = i >> 6, k = i & 63; s_c1t[n * WLD + k] = (_Float16)((k < 43) ? cW1[k * 64 + n] : 0.f); }
    for (int i = tid; i < 64 * 64; i += NPB) { int n = i >> 6, k = i & 63; s_c2t[n * WLD + k] = (_Float16)cW2[k * 64 + n]; }
    for (int i = tid; i < 16 * 64; i += NPB) { int n = i >> 6, k = i & 63; s_c3t[n * WLD + k] = (_Float16)((n < 3) ? cW3[k * 3 + n] : 0.f); }
    if (tid < 64) { s_b1[tid] = db1[tid]; s_bc1[tid] = cb1[tid]; s_bc2[tid] = cb2[tid]; }
    if (tid < 16) { s_b2[tid] = db2[tid]; s_bc3[tid] = (tid < 3) ? cb3[tid] : 0.f; }

    // ---------------- Phase 2: hash encode + positional encode (1 thread = 1 point) ----------------
    {
        const int p = p0 + tid;
        const float px = x[3 * p + 0], py = x[3 * p + 1], pz = x[3 * p + 2];
        const float xs0 = px * (1.f / 3.f), xs1 = py * (1.f / 3.f), xs2 = pz * (1.f / 3.f);
        const bool m = (fabsf(xs0) < 0.5f) && (fabsf(xs1) < 0.5f) && (fabsf(xs2) < 0.5f);
        s_mask[tid] = m ? 1 : 0;
        const float u0 = xs0 + 0.5f, u1 = xs1 + 0.5f, u2 = xs2 + 0.5f;

        _Float16* arow = &s_act[tid * LDA];
        for (int l = 0; l < 16; ++l) {
            const float Nf = c_NL[l];
            const float sx = u0 * Nf, sy = u1 * Nf, sz = u2 * Nf;
            const float fx = floorf(sx), fy = floorf(sy), fz = floorf(sz);
            const float wx = sx - fx, wy = sy - fy, wz = sz - fz;
            const unsigned ix0 = (unsigned)(int)fx, iy0 = (unsigned)(int)fy, iz0 = (unsigned)(int)fz;
            const unsigned ix1 = (unsigned)(int)ceilf(sx), iy1 = (unsigned)(int)ceilf(sy), iz1 = (unsigned)(int)ceilf(sz);
            const float2* tb = (const float2*)(tables + (size_t)l * T_HASH * 2u);
            float f0 = 0.f, f1 = 0.f;
#pragma unroll
            for (int v = 0; v < 8; ++v) {
                const unsigned hx = (v & 1) ? ix1 : ix0;
                const unsigned hy = (v & 2) ? iy1 : iy0;
                const unsigned hz = (v & 4) ? iz1 : iz0;
                const unsigned h = hx ^ (hy * 2654435761u) ^ (hz * 805459861u);
                const float2 val = tb[h & (T_HASH - 1u)];
                const float wv = ((v & 1) ? wx : 1.f - wx) *
                                 ((v & 2) ? wy : 1.f - wy) *
                                 ((v & 4) ? wz : 1.f - wz);
                f0 = fmaf(wv, val.x, f0);
                f1 = fmaf(wv, val.y, f1);
            }
            arow[2 * l + 0] = (_Float16)f0;
            arow[2 * l + 1] = (_Float16)f1;
        }

        // positional encoding -> cin cols 16..42 ; pad 43..63 with zeros
        const float dx = dir[3 * p + 0], dy = dir[3 * p + 1], dz = dir[3 * p + 2];
        _Float16* crow = &s_cin[tid * LDA];
        crow[16] = (_Float16)dx; crow[17] = (_Float16)dy; crow[18] = (_Float16)dz;
#pragma unroll
        for (int j = 0; j < 4; ++j) {
            const float sc = (float)(1 << j);
            crow[19 + 6 * j + 0] = (_Float16)__sinf(sc * dx);
            crow[19 + 6 * j + 1] = (_Float16)__sinf(sc * dy);
            crow[19 + 6 * j + 2] = (_Float16)__sinf(sc * dz);
            crow[22 + 6 * j + 0] = (_Float16)__cosf(sc * dx);
            crow[22 + 6 * j + 1] = (_Float16)__cosf(sc * dy);
            crow[22 + 6 * j + 2] = (_Float16)__cosf(sc * dz);
        }
#pragma unroll
        for (int k = 43; k < 64; ++k) crow[k] = (_Float16)0.f;
    }

    __syncthreads();

    // ---------------- Phase 3: fused MLPs via WMMA (each wave: two 16-point tiles) ----------------
    const int wid  = tid >> 5;
    const int lane = tid & 31;
    const int ln   = lane & 15;
    const int mb   = (lane >> 4) * 8;

    for (int t = wid * 2; t < wid * 2 + 2; ++t) {
        _Float16* A = &s_act[t * 16 * LDA];
        _Float16* C = &s_cin[t * 16 * LDA];

        // density layer 1: relu(feat[16x32] @ dW1[32x64] + db1)  (in-place into A)
        {
            v16h a = load_a_frag(A, LDA, 0);  // read before any store
#pragma unroll
            for (int j = 0; j < 4; ++j) {
                v16h b = load_b_frag(s_w1t, W1LD, 16 * j, 0);
                v8f c = bias_frag(s_b1, 16 * j);
                c = wmma32(a, b, c);
                store_d_frag(A, LDA, 16 * j, c, true);
            }
        }
        // density layer 2: hden = h1[16x64] @ dW2[64x16] + db2 -> cin cols 0..15 + sigma
        {
            v16h a0 = load_a_frag(A, LDA, 0);
            v16h a1 = load_a_frag(A, LDA, 32);
            v16h b0 = load_b_frag(s_w2t, WLD, 0, 0);
            v16h b1 = load_b_frag(s_w2t, WLD, 0, 32);
            v8f c = bias_frag(s_b2, 0);
            c = wmma32(a0, b0, c);
            c = wmma32(a1, b1, c);
            store_d_frag(C, LDA, 0, c, false);
            if (ln == 0) {  // column 0 -> log_sigma; lanes 0 & 16 cover rows 0-7 / 8-15
#pragma unroll
                for (int r = 0; r < 8; ++r) {
                    const int pt = t * 16 + mb + r;
                    s_sig[pt] = s_mask[pt] ? expf(c[r]) : 0.f;
                }
            }
        }
        // color layer 1: relu(cin[16x64] @ cW1[64x64] + cb1) -> A
        {
            v16h a0 = load_a_frag(C, LDA, 0);
            v16h a1 = load_a_frag(C, LDA, 32);
#pragma unroll
            for (int j = 0; j < 4; ++j) {
                v16h b0 = load_b_frag(s_c1t, WLD, 16 * j, 0);
                v16h b1 = load_b_frag(s_c1t, WLD, 16 * j, 32);
                v8f c = bias_frag(s_bc1, 16 * j);
                c = wmma32(a0, b0, c);
                c = wmma32(a1, b1, c);
                store_d_frag(A, LDA, 16 * j, c, true);
            }
        }
        // color layer 2: relu(h[16x64] @ cW2[64x64] + cb2) (in-place in A)
        {
            v16h a0 = load_a_frag(A, LDA, 0);
            v16h a1 = load_a_frag(A, LDA, 32);
#pragma unroll
            for (int j = 0; j < 4; ++j) {
                v16h b0 = load_b_frag(s_c2t, WLD, 16 * j, 0);
                v16h b1 = load_b_frag(s_c2t, WLD, 16 * j, 32);
                v8f c = bias_frag(s_bc2, 16 * j);
                c = wmma32(a0, b0, c);
                c = wmma32(a1, b1, c);
                store_d_frag(A, LDA, 16 * j, c, true);
            }
        }
        // color layer 3: sigmoid(h[16x64] @ cW3[64x3(pad16)] + cb3) -> staged color
        {
            v16h a0 = load_a_frag(A, LDA, 0);
            v16h a1 = load_a_frag(A, LDA, 32);
            v16h b0 = load_b_frag(s_c3t, WLD, 0, 0);
            v16h b1 = load_b_frag(s_c3t, WLD, 0, 32);
            v8f c = bias_frag(s_bc3, 0);
            c = wmma32(a0, b0, c);
            c = wmma32(a1, b1, c);
            if (ln < 3) {
#pragma unroll
                for (int r = 0; r < 8; ++r) {
                    const int pt = t * 16 + mb + r;
                    const float v = 1.f / (1.f + expf(-c[r]));
                    s_col[pt * 3 + ln] = s_mask[pt] ? v : 0.f;
                }
            }
        }
    }

    __syncthreads();

    // ---------------- Phase 4: coalesced output writeback ----------------
    out_sigma[p0 + tid] = s_sig[tid];
#pragma unroll
    for (int j = 0; j < 3; ++j)
        out_color[(size_t)(p0 + tid) * 3 + j] = s_col[tid * 3 + j];
}

extern "C" void kernel_launch(void* const* d_in, const int* in_sizes, int n_in,
                              void* d_out, int out_size, void* d_ws, size_t ws_size,
                              hipStream_t stream) {
    const float* x   = (const float*)d_in[0];
    const float* dir = (const float*)d_in[1];
    const float* tbl = (const float*)d_in[2];
    const float* dW1 = (const float*)d_in[3];
    const float* db1 = (const float*)d_in[4];
    const float* dW2 = (const float*)d_in[5];
    const float* db2 = (const float*)d_in[6];
    const float* cW1 = (const float*)d_in[7];
    const float* cb1 = (const float*)d_in[8];
    const float* cW2 = (const float*)d_in[9];
    const float* cb2 = (const float*)d_in[10];
    const float* cW3 = (const float*)d_in[11];
    const float* cb3 = (const float*)d_in[12];

    const int n = in_sizes[0] / 3;            // 1,048,576 points (multiple of 256)
    float* out_color = (float*)d_out;         // [n,3] then sigma [n]
    float* out_sigma = out_color + (size_t)n * 3;

    // allow >64KB dynamic LDS (gfx1250: up to 320KB per workgroup)
    (void)hipFuncSetAttribute((const void*)ngp_fused_kernel,
                              hipFuncAttributeMaxDynamicSharedMemorySize,
                              SMEM_BYTES);

    const int blocks = n / NPB;
    hipLaunchKernelGGL(ngp_fused_kernel, dim3(blocks), dim3(NPB), SMEM_BYTES, stream,
                       x, dir, tbl, dW1, db1, dW2, db2,
                       cW1, cb1, cW2, cb2, cW3, cb3,
                       out_color, out_sigma);
}
